// Mamba2DiffusionBlock_70497593196672
// MI455X (gfx1250) — matrix operand order repaired and checked by reading, not compiled, not run on hardware
//
#include <hip/hip_runtime.h>
#include <math.h>

// ---------------- problem constants ----------------
#define DM     768
#define DSTATE 128
#define HD     64
#define NH     24
#define DI     1536
#define CONVD  1792
#define NPROJ  3352
#define NPROJP 3456      // padded to multiple of 128 for the big GEMM tile
#define LSEQ   2048
#define BB     2
#define BLROWS (BB*LSEQ) // 4096
#define NCHK   32
#define CHL    64

typedef unsigned short u16;
typedef __attribute__((ext_vector_type(16))) __bf16          v16bf;
typedef __attribute__((ext_vector_type(16))) unsigned short  v16u;
typedef __attribute__((ext_vector_type(8)))  float           v8f;

// ---------------- helpers ----------------
__device__ __forceinline__ u16 f2bf(float x) {
  unsigned u = __float_as_uint(x);
  u += 0x7FFFu + ((u >> 16) & 1u);      // round to nearest even
  return (u16)(u >> 16);
}
__device__ __forceinline__ float bf2f(u16 h) {
  return __uint_as_float(((unsigned)h) << 16);
}
__device__ __forceinline__ float siluf(float v) { return v / (1.0f + expf(-v)); }

// A fragment: 16x32 bf16, ISA 7.12.2 layout. p -> element [row0][k0], ld in elems.
__device__ __forceinline__ v16bf fragA16(const u16* p, int ld) {
  int lane = threadIdx.x & 31;
  const u16* r = p + (lane & 15) * ld + ((lane >> 4) << 3);
  v16u f;
#pragma unroll
  for (int i = 0; i < 8; ++i) f[i] = r[i];
#pragma unroll
  for (int i = 0; i < 8; ++i) f[8 + i] = r[16 + i];
  return __builtin_bit_cast(v16bf, f);
}
// B fragment: 32x16 bf16 from W stored [N][K] (K contiguous). lanes0-15: K 0..7,16..23, lanes16-31: K 8..15,24..31.
__device__ __forceinline__ v16bf fragB16(const u16* p, int ld) {
  int lane = threadIdx.x & 31;
  const u16* r = p + (lane & 15) * ld + ((lane >> 4) << 3);
  v16u f;
#pragma unroll
  for (int i = 0; i < 8; ++i) f[i] = r[i];
#pragma unroll
  for (int i = 0; i < 8; ++i) f[8 + i] = r[16 + i];
  return __builtin_bit_cast(v16bf, f);
}
__device__ __forceinline__ v8f wmma_bf(v16bf a, v16bf b, v8f c) {
  return __builtin_amdgcn_wmma_f32_16x16x32_bf16(false, a, false, b, (short)0, c, false, false);
}
__device__ __forceinline__ v16bf scale_fragA(v16bf a, float s) {
  v16u u = __builtin_bit_cast(v16u, a);
#pragma unroll
  for (int i = 0; i < 16; ++i) u[i] = f2bf(bf2f(u[i]) * s);
  return __builtin_bit_cast(v16bf, u);
}

// ---------------- generic WMMA GEMM: C[M,N] = A[M,K](bf16) * W[N,K](bf16)^T (+bias)(+res) --------
// 128x128 block tile, 256 threads = 8 waves (4 M x 2 N); each wave: 32x64 output = 8 accumulators.
// Per K-step(32): 8 WMMAs vs 6 LDS fragment loads per wave -> WMMA-dense inner loop.
__global__ __launch_bounds__(256) void k_gemm(
    const u16* __restrict__ A, const u16* __restrict__ W,
    const float* __restrict__ bias, const float* __restrict__ res,
    float* __restrict__ C, int M, int N, int K, int lda, int ldw, int ldc) {
  __shared__ __align__(16) u16 sA[128][48];
  __shared__ __align__(16) u16 sW[128][48];
  int bn = blockIdx.x * 128, bm = blockIdx.y * 128;
  if (bm >= M || bn >= N) return;
  int tid = threadIdx.x;
  int w = tid >> 5, wm = w >> 1, wn = w & 1;
  int lrow = tid >> 2, lseg = (tid & 3) * 8;
  v8f acc[2][4] = {};
  for (int k0 = 0; k0 < K; k0 += 32) {
#pragma unroll
    for (int r = 0; r < 2; ++r) {
      int row = lrow + r * 64;
      *(uint4*)&sA[row][lseg] = *(const uint4*)(A + (long)(bm + row) * lda + k0 + lseg);
      *(uint4*)&sW[row][lseg] = *(const uint4*)(W + (long)(bn + row) * ldw + k0 + lseg);
    }
    __syncthreads();
    v16bf a0 = fragA16(&sA[wm * 32][0], 48);
    v16bf a1 = fragA16(&sA[wm * 32 + 16][0], 48);
#pragma unroll
    for (int j = 0; j < 4; ++j) {
      v16bf bj = fragB16(&sW[wn * 64 + j * 16][0], 48);
      acc[0][j] = wmma_bf(a0, bj, acc[0][j]);
      acc[1][j] = wmma_bf(a1, bj, acc[1][j]);
    }
    __syncthreads();
  }
  int lane = tid & 31;
#pragma unroll
  for (int si = 0; si < 2; ++si) {
    int rb = bm + wm * 32 + si * 16 + ((lane >> 4) << 3);
#pragma unroll
    for (int sj = 0; sj < 4; ++sj) {
      int col = bn + wn * 64 + sj * 16 + (lane & 15);
#pragma unroll
      for (int i = 0; i < 8; ++i) {
        long o = (long)(rb + i) * ldc + col;
        float v = acc[si][sj][i];
        if (bias) v += bias[col];
        if (res)  v += res[o];
        C[o] = v;
      }
    }
  }
}

// ---------------- scale/shift: (emb @ W^T + b) ----------------
__global__ void k_scale_shift(const float* __restrict__ emb,
                              const float* __restrict__ sW, const float* __restrict__ sb,
                              const float* __restrict__ hW, const float* __restrict__ hb,
                              float* __restrict__ scale, float* __restrict__ shift) {
  int id = blockIdx.x * 256 + threadIdx.x;
  if (id >= 2 * BB * DM) return;
  int which = id / (BB * DM);
  int rem = id % (BB * DM);
  int b = rem / DM, d = rem % DM;
  const float* Wm = which ? hW : sW;
  float acc = (which ? hb : sb)[d];
  const float* e = emb + b * DM;
  const float* wr = Wm + (long)d * DM;
  for (int k = 0; k < DM; ++k) acc += e[k] * wr[k];
  (which ? shift : scale)[b * DM + d] = acc;
}

// ---------------- LayerNorm + modulation -> bf16 ----------------
__global__ __launch_bounds__(256) void k_ln_mod(const float* __restrict__ x,
    const float* __restrict__ g, const float* __restrict__ be,
    const float* __restrict__ scale, const float* __restrict__ shift,
    u16* __restrict__ xn_bf) {
  int t = blockIdx.x, b = blockIdx.y;
  const float* row = x + ((long)b * LSEQ + t) * DM;
  __shared__ float rs[256], rq[256];
  float v[3], s = 0.f, q = 0.f;
#pragma unroll
  for (int j = 0; j < 3; ++j) {
    v[j] = row[threadIdx.x + j * 256];
    s += v[j]; q += v[j] * v[j];
  }
  rs[threadIdx.x] = s; rq[threadIdx.x] = q; __syncthreads();
  for (int o = 128; o > 0; o >>= 1) {
    if (threadIdx.x < o) { rs[threadIdx.x] += rs[threadIdx.x + o]; rq[threadIdx.x] += rq[threadIdx.x + o]; }
    __syncthreads();
  }
  float mu = rs[0] * (1.0f / DM);
  float var = rq[0] * (1.0f / DM) - mu * mu;
  float inv = rsqrtf(var + 1e-5f);
  u16* orow = xn_bf + ((long)b * LSEQ + t) * DM;
#pragma unroll
  for (int j = 0; j < 3; ++j) {
    int i = threadIdx.x + j * 256;
    float xv = (v[j] - mu) * inv * g[i] + be[i];
    xv = xv * (1.0f + scale[b * DM + i]) + shift[b * DM + i];
    orow[i] = f2bf(xv);
  }
}

// ---------------- f32 -> bf16 convert (with optional zero pad tail) ----------------
__global__ void k_convert(const float* __restrict__ src, u16* __restrict__ dst,
                          long n_valid, long n_total) {
  long i = (long)blockIdx.x * 256 + threadIdx.x;
  if (i < n_total) dst[i] = (i < n_valid) ? f2bf(src[i]) : (u16)0;
}

// ---------------- padded-row f32 -> bf16 convert for weights: dst[prow][K], zero pad rows ------
__global__ void k_convert_rows(const float* __restrict__ src, u16* __restrict__ dst,
                               long rows, long cols, long prows) {
  long i = (long)blockIdx.x * 256 + threadIdx.x;
  if (i < prows * cols) dst[i] = (i < rows * cols) ? f2bf(src[i]) : (u16)0;
}

// ---------------- causal depthwise conv (k=4) + SiLU, logical time (rev flips reads) ----------------
__global__ void k_conv(const float* __restrict__ zx, const float* __restrict__ cw,
                       const float* __restrict__ cb, float* __restrict__ xBC, int rev) {
  int ch = blockIdx.x * 256 + threadIdx.x;
  int t = blockIdx.y, b = blockIdx.z;
  if (ch >= CONVD) return;
  float acc = cb[ch];
  const float* wch = cw + ch * 4;
#pragma unroll
  for (int k = 0; k < 4; ++k) {
    int tau = t - 3 + k;
    if (tau >= 0) {
      int tp = rev ? (LSEQ - 1 - tau) : tau;
      acc += wch[k] * zx[((long)b * LSEQ + tp) * NPROJP + DI + ch];
    }
  }
  xBC[((long)b * LSEQ + t) * CONVD + ch] = siluf(acc);
}

// ---------------- dt = softplus(dt_raw + bias); dA = dt * (-exp(A_log)) ----------------
__global__ void k_dt(const float* __restrict__ zx, const float* __restrict__ dt_bias,
                     const float* __restrict__ A_log, float* __restrict__ dt_s,
                     float* __restrict__ dA, int rev) {
  int id = blockIdx.x * 256 + threadIdx.x;
  if (id >= BB * NH * LSEQ) return;
  int t = id % LSEQ;
  int h = (id / LSEQ) % NH;
  int b = id / (LSEQ * NH);
  int tp = rev ? (LSEQ - 1 - t) : t;
  float v = zx[((long)b * LSEQ + tp) * NPROJP + DI + CONVD + h] + dt_bias[h];
  float dtv = (v > 20.f) ? v : log1pf(expf(v));
  long o = ((long)b * NH + h) * LSEQ + t;
  dt_s[o] = dtv;
  dA[o] = dtv * (-expf(A_log[h]));
}

// ---------------- per-chunk inclusive cumsum + decays ----------------
__global__ void k_cumsum(const float* __restrict__ dA, float* __restrict__ Acs,
                         float* __restrict__ dec, float* __restrict__ cdec) {
  int id = blockIdx.x * 256 + threadIdx.x;
  if (id >= BB * NH * NCHK) return;
  int c = id % NCHK;
  int bh = id / NCHK;
  long base = (long)bh * LSEQ + c * CHL;
  float a = 0.f;
  for (int l = 0; l < CHL; ++l) { a += dA[base + l]; Acs[base + l] = a; }
  cdec[id] = expf(a);
  for (int l = 0; l < CHL; ++l) dec[base + l] = expf(a - Acs[base + l]);
}

// ---------------- build bf16 operands + init Y = x*D ----------------
__global__ void k_operands(const float* __restrict__ xBC, const float* __restrict__ dt_s,
                           const float* __restrict__ dec, const float* __restrict__ Dp,
                           u16* __restrict__ XsT, u16* __restrict__ XsD,
                           u16* __restrict__ Cm, u16* __restrict__ Bm, u16* __restrict__ BmT,
                           float* __restrict__ Y) {
  int ch = blockIdx.x * 256 + threadIdx.x;
  int t = blockIdx.y, b = blockIdx.z;
  if (ch >= CONVD) return;
  float v = xBC[((long)b * LSEQ + t) * CONVD + ch];
  if (ch < DI) {
    int h = ch >> 6, p = ch & 63;
    long bh = (long)b * NH + h;
    float dtv = dt_s[bh * LSEQ + t];
    float xs = v * dtv;
    XsT[(bh * HD + p) * LSEQ + t] = f2bf(xs);
    XsD[(bh * HD + p) * LSEQ + t] = f2bf(xs * dec[bh * LSEQ + t]);
    Y[((long)b * LSEQ + t) * DI + ch] = v * Dp[h];
  } else if (ch < DI + DSTATE) {
    int n = ch - DI;
    Bm[((long)b * LSEQ + t) * DSTATE + n] = f2bf(v);
    BmT[((long)b * DSTATE + n) * LSEQ + t] = f2bf(v);
  } else {
    int n = ch - DI - DSTATE;
    Cm[((long)b * LSEQ + t) * DSTATE + n] = f2bf(v);
  }
}

// ---------------- chunk states: st[p,n] = sum_l XsDec_T[p,l] * BmT[n,l]  (K=64) ----------------
// grid (NCHK, NH, BB); 8 waves: wp=w>>1 rows, wn2=w&1 half of 128 cols; 4 accums each.
__global__ __launch_bounds__(256) void k_st(const u16* __restrict__ XsD, const u16* __restrict__ BmT,
                                            float* __restrict__ st) {
  int c = blockIdx.x, h = blockIdx.y, b = blockIdx.z;
  int w = threadIdx.x >> 5, wp = w >> 1, wn2 = w & 1;
  v8f acc[4] = {};
  const u16* Abase = XsD + ((long)(b * NH + h) * HD + wp * 16) * LSEQ + c * CHL;
  const u16* Wbase = BmT + ((long)b * DSTATE + wn2 * 64) * LSEQ + c * CHL;
#pragma unroll
  for (int k0 = 0; k0 < CHL; k0 += 32) {
    v16bf a = fragA16(Abase + k0, LSEQ);
#pragma unroll
    for (int j = 0; j < 4; ++j) {
      v16bf bj = fragB16(Wbase + (long)(j * 16) * LSEQ + k0, LSEQ);
      acc[j] = wmma_bf(a, bj, acc[j]);
    }
  }
  int lane = threadIdx.x & 31;
  int rb = wp * 16 + ((lane >> 4) << 3);
  long base = ((long)(b * NCHK + c) * NH + h) * HD;
#pragma unroll
  for (int j = 0; j < 4; ++j) {
    int col = wn2 * 64 + j * 16 + (lane & 15);
#pragma unroll
    for (int i = 0; i < 8; ++i)
      st[(base + rb + i) * DSTATE + col] = acc[j][i];
  }
}

// ---------------- inter-chunk recurrence (in place): st[c] <- prev state before chunk c ----------------
__global__ void k_scan(float* __restrict__ st, const float* __restrict__ cdec) {
  long id = (long)blockIdx.x * 256 + threadIdx.x;
  if (id >= (long)BB * NH * HD * DSTATE) return;
  int n = (int)(id % DSTATE);
  long r = id / DSTATE;
  int p = (int)(r % HD); r /= HD;
  int h = (int)(r % NH);
  int b = (int)(r / NH);
  const float* cd = cdec + (b * NH + h) * NCHK;
  float carry = 0.f;
  for (int c = 0; c < NCHK; ++c) {
    long o = (((long)(b * NCHK + c) * NH + h) * HD + p) * DSTATE + n;
    float s = st[o];
    st[o] = carry;
    carry = carry * cd[c] + s;
  }
}

// ---------------- fused per-(b,c,h): G=C*B^T -> M=tril(G*Lm) -> Y += M*Xs + diag(e^Acs)*C*prev^T ----
__global__ __launch_bounds__(256) void k_ydyo(const u16* __restrict__ Cm, const u16* __restrict__ Bm,
                                              const u16* __restrict__ XsT, const float* __restrict__ st,
                                              const float* __restrict__ Acs, float* __restrict__ Y) {
  int c = blockIdx.x, h = blockIdx.y, b = blockIdx.z;
  int w = threadIdx.x >> 5, wm = w >> 1, wn = w & 1;
  int lane = threadIdx.x & 31;
  __shared__ float sAcs[CHL];
  __shared__ __align__(16) u16 sM[64][72];
  __shared__ __align__(16) u16 sPrev[64][136];

  if (threadIdx.x < CHL)
    sAcs[threadIdx.x] = Acs[((long)b * NH + h) * LSEQ + c * CHL + threadIdx.x];

  // stage 1: S = C (64x128) @ B^T  -> registers
  v8f s0 = {}, s1 = {};
  const u16* Cb = Cm + ((long)b * LSEQ + c * CHL + wm * 16) * DSTATE;
  const u16* Bb0 = Bm + ((long)b * LSEQ + c * CHL + wn * 32) * DSTATE;
  const u16* Bb1 = Bb0 + (long)16 * DSTATE;
#pragma unroll
  for (int k0 = 0; k0 < DSTATE; k0 += 32) {
    v16bf a = fragA16(Cb + k0, DSTATE);
    s0 = wmma_bf(a, fragB16(Bb0 + k0, DSTATE), s0);
    s1 = wmma_bf(a, fragB16(Bb1 + k0, DSTATE), s1);
  }
  // stage prev (f32 -> bf16) into LDS
  const float* stp = st + ((long)(b * NCHK + c) * NH + h) * HD * DSTATE;
  for (int e = threadIdx.x; e < HD * DSTATE; e += 256)
    sPrev[e >> 7][e & 127] = f2bf(stp[e]);
  __syncthreads();

  // write M = tril(S * exp(Acs[l]-Acs[s])) as bf16
  {
    int rb = wm * 16 + ((lane >> 4) << 3);
    int cb = wn * 32 + (lane & 15);
#pragma unroll
    for (int i = 0; i < 8; ++i) {
      int l = rb + i;
      int s = cb;
      float v0 = (l >= s) ? s0[i] * expf(sAcs[l] - sAcs[s]) : 0.f;
      int s2 = cb + 16;
      float v1 = (l >= s2) ? s1[i] * expf(sAcs[l] - sAcs[s2]) : 0.f;
      sM[l][s] = f2bf(v0);
      sM[l][s2] = f2bf(v1);
    }
  }
  __syncthreads();

  // stage 2: Yd + Yo
  v8f y0 = {}, y1 = {};
  const u16* Xb0 = XsT + ((long)(b * NH + h) * HD + wn * 32) * LSEQ + c * CHL;
  const u16* Xb1 = Xb0 + (long)16 * LSEQ;
#pragma unroll
  for (int k0 = 0; k0 < CHL; k0 += 32) {           // Yd: K = 64 over s
    v16bf a = fragA16(&sM[wm * 16][k0], 72);
    y0 = wmma_bf(a, fragB16(Xb0 + k0, LSEQ), y0);
    y1 = wmma_bf(a, fragB16(Xb1 + k0, LSEQ), y1);
  }
  float rowScale = expf(sAcs[wm * 16 + (lane & 15)]);
#pragma unroll
  for (int k0 = 0; k0 < DSTATE; k0 += 32) {        // Yo: K = 128 over n
    v16bf a = scale_fragA(fragA16(Cb + k0, DSTATE), rowScale);
    y0 = wmma_bf(a, fragB16(&sPrev[wn * 32][k0], 136), y0);
    y1 = wmma_bf(a, fragB16(&sPrev[wn * 32 + 16][k0], 136), y1);
  }
  // accumulate into Y (disjoint tiles per block)
  {
    int rb = wm * 16 + ((lane >> 4) << 3);
    int cb = wn * 32 + (lane & 15);
#pragma unroll
    for (int i = 0; i < 8; ++i) {
      long o = ((long)b * LSEQ + c * CHL + rb + i) * DI + h * HD;
      Y[o + cb] += y0[i];
      Y[o + cb + 16] += y1[i];
    }
  }
}

// ---------------- gate with silu(z), RMSNorm, write bf16 at PHYSICAL time (handles flip) ----------------
__global__ __launch_bounds__(256) void k_gate_norm(const float* __restrict__ Y,
    const float* __restrict__ zx, const float* __restrict__ nw,
    u16* __restrict__ ybf, int rev) {
  int t = blockIdx.x, b = blockIdx.y;
  int tp = rev ? (LSEQ - 1 - t) : t;
  const float* yrow = Y + ((long)b * LSEQ + t) * DI;
  const float* zrow = zx + ((long)b * LSEQ + tp) * NPROJP;
  float vals[6], ss = 0.f;
#pragma unroll
  for (int j = 0; j < 6; ++j) {
    int i = threadIdx.x + j * 256;
    float yv = yrow[i] * siluf(zrow[i]);
    vals[j] = yv; ss += yv * yv;
  }
  __shared__ float rs[256];
  rs[threadIdx.x] = ss; __syncthreads();
  for (int o = 128; o > 0; o >>= 1) {
    if (threadIdx.x < o) rs[threadIdx.x] += rs[threadIdx.x + o];
    __syncthreads();
  }
  float inv = rsqrtf(rs[0] * (1.0f / DI) + 1e-5f);
  u16* orow = ybf + ((long)b * LSEQ + tp) * DI;
#pragma unroll
  for (int j = 0; j < 6; ++j) {
    int i = threadIdx.x + j * 256;
    orow[i] = f2bf(vals[j] * inv * nw[i]);
  }
}

// ---------------- host ----------------
extern "C" void kernel_launch(void* const* d_in, const int* in_sizes, int n_in,
                              void* d_out, int out_size, void* d_ws, size_t ws_size,
                              hipStream_t stream) {
  (void)in_sizes; (void)n_in; (void)out_size; (void)ws_size;
  const float* x     = (const float*)d_in[0];
  const float* emb   = (const float*)d_in[1];
  const float* ln_g  = (const float*)d_in[2];
  const float* ln_b  = (const float*)d_in[3];
  const float* scW   = (const float*)d_in[4];
  const float* scb   = (const float*)d_in[5];
  const float* shW   = (const float*)d_in[6];
  const float* shb   = (const float*)d_in[7];
  const float* out_W = (const float*)d_in[24];
  const float* out_b = (const float*)d_in[25];

  char* ws = (char*)d_ws;
  size_t off = 0;
  auto take = [&](size_t bytes) -> void* {
    size_t a = (off + 255) & ~(size_t)255;
    off = a + bytes;
    return (void*)(ws + a);
  };
  u16*   xn_bf  = (u16*)  take((size_t)BLROWS * DM * 2);
  u16*   win_bf = (u16*)  take((size_t)NPROJP * DM * 2);
  u16*   wout_bf= (u16*)  take((size_t)DM * DI * 2);
  u16*   wfin_bf= (u16*)  take((size_t)DM * DI * 2);
  float* scale  = (float*)take((size_t)BB * DM * 4);
  float* shift  = (float*)take((size_t)BB * DM * 4);
  float* zx     = (float*)take((size_t)BLROWS * NPROJP * 4);
  float* xBC    = (float*)take((size_t)BLROWS * CONVD * 4);
  float* dt_s   = (float*)take((size_t)BB * NH * LSEQ * 4);
  float* dAb    = (float*)take((size_t)BB * NH * LSEQ * 4);
  float* Acs    = (float*)take((size_t)BB * NH * LSEQ * 4);
  float* dec    = (float*)take((size_t)BB * NH * LSEQ * 4);
  float* cdec   = (float*)take((size_t)BB * NH * NCHK * 4);
  u16*   Cmb    = (u16*)  take((size_t)BLROWS * DSTATE * 2);
  u16*   Bmb    = (u16*)  take((size_t)BLROWS * DSTATE * 2);
  u16*   BmT    = (u16*)  take((size_t)BB * DSTATE * LSEQ * 2);
  u16*   XsT    = (u16*)  take((size_t)BB * NH * HD * LSEQ * 2);
  u16*   XsD    = (u16*)  take((size_t)BB * NH * HD * LSEQ * 2);
  float* st     = (float*)take((size_t)BB * NCHK * NH * HD * DSTATE * 4);
  float* Y      = (float*)take((size_t)BLROWS * DI * 4);
  float* cat    = (float*)take((size_t)BLROWS * DI * 4);
  u16*   ybf    = (u16*)st;   // alias: st dead before gate_norm writes ybf
  u16*   cat_bf = (u16*)zx;   // alias: zx dead after both direction passes

  dim3 blk(256);
  auto cvt = [&](const float* s, u16* d, long nv, long nt) {
    k_convert<<<dim3((unsigned)((nt + 255) / 256)), blk, 0, stream>>>(s, d, nv, nt);
  };

  cvt(out_W, wfin_bf, (long)DM * DI, (long)DM * DI);
  k_scale_shift<<<dim3((2 * BB * DM + 255) / 256), blk, 0, stream>>>(emb, scW, scb, shW, shb, scale, shift);
  k_ln_mod<<<dim3(LSEQ, BB), blk, 0, stream>>>(x, ln_g, ln_b, scale, shift, xn_bf);

  for (int dir = 0; dir < 2; ++dir) {
    int base = 8 + dir * 8;
    const float* Win     = (const float*)d_in[base + 0];
    const float* conv_w  = (const float*)d_in[base + 1];
    const float* conv_b  = (const float*)d_in[base + 2];
    const float* dt_bias = (const float*)d_in[base + 3];
    const float* A_log   = (const float*)d_in[base + 4];
    const float* Dp      = (const float*)d_in[base + 5];
    const float* norm_w  = (const float*)d_in[base + 6];
    const float* Wout    = (const float*)d_in[base + 7];

    cvt(Win, win_bf, (long)NPROJ * DM, (long)NPROJP * DM);
    cvt(Wout, wout_bf, (long)DM * DI, (long)DM * DI);

    // in-projection: zx[4096, 3456] = xn_bf @ Win^T   (padded cols produce zeros)
    k_gemm<<<dim3(NPROJP / 128, BLROWS / 128), blk, 0, stream>>>(
        xn_bf, win_bf, nullptr, nullptr, zx, BLROWS, NPROJP, DM, DM, DM, NPROJP);

    k_conv<<<dim3(CONVD / 256, LSEQ, BB), blk, 0, stream>>>(zx, conv_w, conv_b, xBC, dir);
    k_dt<<<dim3((BB * NH * LSEQ + 255) / 256), blk, 0, stream>>>(zx, dt_bias, A_log, dt_s, dAb, dir);
    k_cumsum<<<dim3((BB * NH * NCHK + 255) / 256), blk, 0, stream>>>(dAb, Acs, dec, cdec);
    k_operands<<<dim3(CONVD / 256, LSEQ, BB), blk, 0, stream>>>(
        xBC, dt_s, dec, Dp, XsT, XsD, Cmb, Bmb, BmT, Y);

    k_st<<<dim3(NCHK, NH, BB), blk, 0, stream>>>(XsD, BmT, st);
    k_scan<<<dim3((BB * NH * HD * DSTATE + 255) / 256), blk, 0, stream>>>(st, cdec);
    k_ydyo<<<dim3(NCHK, NH, BB), blk, 0, stream>>>(Cmb, Bmb, XsT, st, Acs, Y);

    k_gate_norm<<<dim3(LSEQ, BB), blk, 0, stream>>>(Y, zx, norm_w, ybf, dir);

    // out-projection -> cat[:, dir*768 : dir*768+768]
    k_gemm<<<dim3(DM / 128, BLROWS / 128), blk, 0, stream>>>(
        ybf, wout_bf, nullptr, nullptr, cat + dir * DM, BLROWS, DM, DI, DI, DI, DI);
  }

  cvt(cat, cat_bf, (long)BLROWS * DI, (long)BLROWS * DI);
  // final: out = x + cat @ out_W^T + out_b
  k_gemm<<<dim3(DM / 128, BLROWS / 128), blk, 0, stream>>>(
      cat_bf, wfin_bf, out_b, x, (float*)d_out, BLROWS, DM, DI, DI, DI, DM);
}
